// GNNArma_13743895347531
// MI455X (gfx1250) — matrix operand order repaired and checked by reading, not compiled, not run on hardware
//
#include <hip/hip_runtime.h>

// ARMA GNN: 3 x (ARMAConv(K=2,T=2) + BN + ReLU), N=100000, E=1600000, F=64.
// Dense GEMMs via v_wmma_f32_16x16x32_f16 (f16 in / f32 acc); edge scatter via
// L2-resident float atomics (25.6MB activation matrix + 12.8MB indices << 192MB L2).

#define FDIM 64
#define KST 2
#define TLAY 2

typedef __attribute__((ext_vector_type(16))) _Float16 v16h;
typedef __attribute__((ext_vector_type(8)))  float    v8f;

// ---------------- utility kernels ----------------

__global__ void zero_f4(float4* p, long n4) {
  long i = (long)blockIdx.x * blockDim.x + threadIdx.x;
  if (i < n4) p[i] = make_float4(0.f, 0.f, 0.f, 0.f);
}

__global__ void deg_accum(const int* __restrict__ col, const float* __restrict__ w,
                          float* deg, int E) {
  int e = blockIdx.x * blockDim.x + threadIdx.x;
  if (e < E) atomicAdd(&deg[col[e]], w[e]);
}

__global__ void norm_k(const int* __restrict__ row, const int* __restrict__ col,
                       const float* __restrict__ w, const float* __restrict__ deg,
                       float* __restrict__ nrm, int E) {
  int e = blockIdx.x * blockDim.x + threadIdx.x;
  if (e < E) {
    float dr = deg[row[e]], dc = deg[col[e]];
    float a = dr > 0.f ? rsqrtf(dr) : 0.f;
    float b = dc > 0.f ? rsqrtf(dc) : 0.f;
    nrm[e] = a * w[e] * b;
  }
}

// convert 64x64 f32 matrices [f][o] into transposed f16 [o][f] (contiguous K per output col)
__global__ void cvt_t(const float* __restrict__ src, _Float16* __restrict__ dst, int nmats) {
  int i = blockIdx.x * blockDim.x + threadIdx.x;
  int tot = nmats * FDIM * FDIM;
  if (i < tot) {
    int m = i >> 12;
    int f = (i >> 6) & 63;
    int o = i & 63;
    dst[m * 4096 + o * 64 + f] = (_Float16)src[i];
  }
}

// ---------------- WMMA GEMM: C = A[N,64] @ W^T[64,64] (+D) (+bias) (relu) ----------------
// Per wave: 32x64 output (two 16-row strips share every B fragment -> 16 WMMAs/wave).
// blockDim=256 (8 waves) -> 256 rows per block. grid.y = K stack.

__device__ __forceinline__ void load_a_frags(const float* A, int m0, int lane,
                                             v16h& a0, v16h& a1) {
  // A fragment 16x32 f16: lane&15 = M row, lane>>4 selects K-halves {0,8} and {16,24}
  int  klo  = (lane & 16) ? 8 : 0;
  const float* Arow = A + (long)(m0 + (lane & 15)) * FDIM;
  #pragma unroll
  for (int i = 0; i < 8; i++) {
    a0[i]     = (_Float16)Arow[klo + i];
    a0[i + 8] = (_Float16)Arow[klo + 16 + i];
    a1[i]     = (_Float16)Arow[32 + klo + i];
    a1[i + 8] = (_Float16)Arow[48 + klo + i];
  }
}

template <bool HAS_D, bool HAS_BIAS, bool RELU>
__device__ __forceinline__ void store_tile(const v8f& c, float* C, const float* D,
                                           const float* bias, int m0, int gc, int lane) {
  int rbase = m0 + ((lane & 16) ? 8 : 0);
  float bv = 0.f;
  if (HAS_BIAS) bv = bias[gc];
  #pragma unroll
  for (int r = 0; r < 8; r++) {
    long idx = (long)(rbase + r) * FDIM + gc;
    float v = c[r];
    if (HAS_D) v += D[idx];
    if (HAS_BIAS) v += bv;           // no dead +0.0 adds in the plain variant
    if (RELU) v = v > 0.f ? v : 0.f;
    C[idx] = v;
  }
}

template <bool HAS_D, bool HAS_BIAS, bool RELU>
__global__ __launch_bounds__(256) void gemm_wmma(
    const float* Abase, const _Float16* __restrict__ Wbase,
    const float* Dbase, const float* __restrict__ biasBase,
    float* Cbase, int nrows,
    long sA, long sW, long sD, long sB, long sC)
{
  int k = blockIdx.y;
  const float*    A    = Abase + (long)k * sA;
  const _Float16* W    = Wbase + (long)k * sW;
  const float*    D    = HAS_D ? Dbase + (long)k * sD : nullptr;
  const float*    bias = HAS_BIAS ? biasBase + (long)k * sB : nullptr;
  float*          C    = Cbase + (long)k * sC;

  int tid  = threadIdx.x;
  int wave = tid >> 5;
  int lane = tid & 31;
  int m0 = (blockIdx.x * 8 + wave) * 32;   // strip0
  int m1 = m0 + 16;                        // strip1
  if (m0 + 16 > nrows) return;
  bool s1 = (m1 + 16 <= nrows);            // wave-uniform -> EXEC all-ones inside branch

  // All A loads issued before any store -> in-place C=A@W is safe per-row/per-wave.
  v16h a0, a1, a2, a3;
  load_a_frags(A, m0, lane, a0, a1);
  if (s1) load_a_frags(A, m1, lane, a2, a3);

  // B fragment 32x16 f16: lane&15 = N col, lane>>4 selects K-halves {0,8} and {16,24}
  int colb = lane & 15;
  int kb   = (lane & 16) ? 8 : 0;

  #pragma unroll
  for (int nt = 0; nt < 4; nt++) {
    int n0 = nt * 16;
    const _Float16* wp0 = W + (long)(n0 + colb) * FDIM + kb; // K chunk 0
    const _Float16* wp1 = wp0 + 32;                          // K chunk 32
    v16h b0, b1;
    #pragma unroll
    for (int i = 0; i < 16; i++) { int kk = i < 8 ? i : i + 8; b0[i] = wp0[kk]; b1[i] = wp1[kk]; }

    v8f c = {};
    c = __builtin_amdgcn_wmma_f32_16x16x32_f16(false, a0, false, b0, (short)0, c, false, false);
    c = __builtin_amdgcn_wmma_f32_16x16x32_f16(false, a1, false, b1, (short)0, c, false, false);
    store_tile<HAS_D, HAS_BIAS, RELU>(c, C, D, bias, m0, n0 + colb, lane);

    if (s1) {
      v8f d = {};
      d = __builtin_amdgcn_wmma_f32_16x16x32_f16(false, a2, false, b0, (short)0, d, false, false);
      d = __builtin_amdgcn_wmma_f32_16x16x32_f16(false, a3, false, b1, (short)0, d, false, false);
      store_tile<HAS_D, HAS_BIAS, RELU>(d, C, D, bias, m1, n0 + colb, lane);
    }
  }
}

// ---------------- edge scatter: tmp[col] += cur[row] * norm ----------------
// blockDim(16,16): 16 lanes x float4 = 64 features per edge, 16 edges per block.

__global__ void scatter_k(const int* __restrict__ row, const int* __restrict__ col,
                          const float* __restrict__ nrm, const float* __restrict__ curBase,
                          float* tmpBase, int E, long sNF)
{
  int k = blockIdx.y;
  const float* cur = curBase + (long)k * sNF;
  float*       tmp = tmpBase + (long)k * sNF;
  int e = blockIdx.x * blockDim.y + threadIdx.y;
  if (e >= E) return;
  int f4 = threadIdx.x;                        // 0..15
  float nv = nrm[e];
  const float4* src = (const float4*)(cur + (long)row[e] * FDIM);
  float4 v = src[f4];
  float* dst = tmp + (long)col[e] * FDIM + f4 * 4;
  atomicAdd(dst + 0, v.x * nv);
  atomicAdd(dst + 1, v.y * nv);
  atomicAdd(dst + 2, v.z * nv);
  atomicAdd(dst + 3, v.w * nv);
}

__global__ void mean2(const float4* __restrict__ a, const float4* __restrict__ b,
                      float4* __restrict__ o, long n4) {
  long i = (long)blockIdx.x * blockDim.x + threadIdx.x;
  if (i < n4) {
    float4 u = a[i], v = b[i];
    o[i] = make_float4(0.5f * (u.x + v.x), 0.5f * (u.y + v.y),
                       0.5f * (u.z + v.z), 0.5f * (u.w + v.w));
  }
}

// ---------------- batchnorm ----------------

__global__ void bn_stats(const float* __restrict__ h, float* stats, int n) {
  int f  = threadIdx.x & 63;
  int rp = blockDim.x >> 6;
  int r0 = blockIdx.x * rp + (threadIdx.x >> 6);
  int stride = gridDim.x * rp;
  float s = 0.f, s2 = 0.f;
  for (int r = r0; r < n; r += stride) {
    float v = h[(long)r * FDIM + f];
    s += v; s2 += v * v;
  }
  atomicAdd(&stats[f], s);
  atomicAdd(&stats[64 + f], s2);
}

__global__ void bn_apply(float* h, const float* __restrict__ stats,
                         const float* __restrict__ g, const float* __restrict__ be,
                         long n, float inv_n) {
  long i = (long)blockIdx.x * blockDim.x + threadIdx.x;
  if (i < n) {
    int f = i & 63;
    float mu  = stats[f] * inv_n;
    float var = stats[64 + f] * inv_n - mu * mu;
    float v = g[f] * (h[i] - mu) * rsqrtf(var + 1e-5f) + be[f];
    h[i] = v > 0.f ? v : 0.f;
  }
}

// ---------------- host orchestration ----------------

static inline int cdiv(long a, long b) { return (int)((a + b - 1) / b); }

extern "C" void kernel_launch(void* const* d_in, const int* in_sizes, int n_in,
                              void* d_out, int out_size, void* d_ws, size_t ws_size,
                              hipStream_t stream) {
  (void)n_in; (void)out_size; (void)ws_size;

  const float* x  = (const float*)d_in[0];
  const int*   ei = (const int*)d_in[1];
  const float* ew = (const float*)d_in[2];
  const float* iwp[3] = {(const float*)d_in[3],  (const float*)d_in[9],  (const float*)d_in[15]};
  const float* wp [3] = {(const float*)d_in[4],  (const float*)d_in[10], (const float*)d_in[16]};
  const float* rwp[3] = {(const float*)d_in[5],  (const float*)d_in[11], (const float*)d_in[17]};
  const float* bp [3] = {(const float*)d_in[6],  (const float*)d_in[12], (const float*)d_in[18]};
  const float* gp [3] = {(const float*)d_in[7],  (const float*)d_in[13], (const float*)d_in[19]};
  const float* bep[3] = {(const float*)d_in[8],  (const float*)d_in[14], (const float*)d_in[20]};

  const int  Nn = in_sizes[0] / FDIM;   // 100000
  const int  E  = in_sizes[2];          // 1600000
  const long NF = (long)Nn * FDIM;

  // workspace layout (floats)
  float* ws    = (float*)d_ws;
  float* deg   = ws;                    // Nn
  float* nrm   = deg + Nn;              // E
  float* XA    = nrm + E;               // NF
  float* XB    = XA + NF;               // NF
  float* CUR   = XB + NF;               // KST*NF
  float* TMPS  = CUR + KST * NF;        // KST*NF
  float* stats = TMPS + KST * NF;       // 128
  _Float16* WH = (_Float16*)(stats + 128); // 24 * 4096 halves (f16, transposed)

  // ---- gcn_norm ----
  zero_f4<<<cdiv(Nn / 4, 256), 256, 0, stream>>>((float4*)deg, Nn / 4);
  deg_accum<<<cdiv(E, 256), 256, 0, stream>>>(ei + E, ew, deg, E);
  norm_k<<<cdiv(E, 256), 256, 0, stream>>>(ei, ei + E, ew, deg, nrm, E);

  // ---- weights -> transposed f16. Per layer block (8 mats): [iw k0,k1][w k0,k1][rw k0t0,k0t1,k1t0,k1t1]
  for (int L = 0; L < 3; L++) {
    _Float16* wl = WH + (long)L * 8 * 4096;
    cvt_t<<<cdiv(2 * 4096, 256), 256, 0, stream>>>(iwp[L], wl,            2);
    cvt_t<<<cdiv(2 * 4096, 256), 256, 0, stream>>>(wp[L],  wl + 2 * 4096, 2);
    cvt_t<<<cdiv(4 * 4096, 256), 256, 0, stream>>>(rwp[L], wl + 4 * 4096, 4);
  }

  dim3 ggrid(cdiv(Nn, 256), KST);
  auto gemm_plain = [&](const float* A, const _Float16* W, float* C, long sA, long sW) {
    gemm_wmma<false, false, false><<<ggrid, 256, 0, stream>>>(
        A, W, nullptr, nullptr, C, Nn, sA, sW, 0, 0, NF);
  };
  auto gemm_dbr = [&](const float* A, const _Float16* W, const float* D, const float* bias,
                      float* C, long sW, long sD, long sB) {
    gemm_wmma<true, true, true><<<ggrid, 256, 0, stream>>>(
        A, W, D, bias, C, Nn, 0, sW, sD, sB, NF);
  };

  dim3 sgrid(cdiv(E, 16), KST);
  dim3 sblk(16, 16);
  auto layer = [&](const float* Xin, float* Xout, const _Float16* WL,
                   const float* bias, const float* gamma, const float* beta) {
    // out0 = Xin @ iw[k]
    gemm_plain(Xin, WL, CUR, 0, 4096);
    // t=0: scatter, then out = relu(scat + Xin @ rw[k,0] + b[k,0])
    zero_f4<<<cdiv(KST * NF / 4, 256), 256, 0, stream>>>((float4*)TMPS, KST * NF / 4);
    scatter_k<<<sgrid, sblk, 0, stream>>>(ei, ei + E, nrm, CUR, TMPS, E, NF);
    gemm_dbr(Xin, WL + 4 * 4096, TMPS, bias, CUR, 2 * 4096, NF, (long)TLAY * FDIM);
    // t=1: out = out @ w[k,0] (in place; row m depends only on row m within owning wave)
    gemm_plain(CUR, WL + 2 * 4096, CUR, NF, 4096);
    zero_f4<<<cdiv(KST * NF / 4, 256), 256, 0, stream>>>((float4*)TMPS, KST * NF / 4);
    scatter_k<<<sgrid, sblk, 0, stream>>>(ei, ei + E, nrm, CUR, TMPS, E, NF);
    gemm_dbr(Xin, WL + 5 * 4096, TMPS, bias + FDIM, CUR, 2 * 4096, NF, (long)TLAY * FDIM);
    // mean over stacks
    mean2<<<cdiv(NF / 4, 256), 256, 0, stream>>>(
        (const float4*)CUR, (const float4*)(CUR + NF), (float4*)Xout, NF / 4);
    // batchnorm + relu (in place on Xout)
    zero_f4<<<1, 32, 0, stream>>>((float4*)stats, 32);
    bn_stats<<<256, 256, 0, stream>>>(Xout, stats, Nn);
    bn_apply<<<cdiv(NF, 256), 256, 0, stream>>>(Xout, stats, gamma, beta, NF, 1.0f / (float)Nn);
  };

  layer(x,  XA,            WH,              bp[0], gp[0], bep[0]);
  layer(XA, XB,            WH + 8 * 4096,   bp[1], gp[1], bep[1]);
  layer(XB, (float*)d_out, WH + 16 * 4096,  bp[2], gp[2], bep[2]);
}
